// BAGCA_49503793054203
// MI455X (gfx1250) — matrix operand order repaired and checked by reading, not compiled, not run on hardware
//
#include <hip/hip_runtime.h>
#include <hip/hip_bf16.h>

#define B_    32
#define DL_   290
#define PL_   1000
#define H_    256
#define HEADS_ 8
#define DK_   32
#define DLP_  320      // DL padded to multiple of 32
#define PLP_  1024     // PL padded to multiple of 32 (and power of two)
#define NROW_ 1290     // DL + PL
#define INV_SQRT_DK 0.1767766952966369f

typedef __attribute__((ext_vector_type(16))) __bf16 v16bf;
typedef __attribute__((ext_vector_type(8)))  float  v8f;
typedef __attribute__((ext_vector_type(4)))  int    v4i;

union PackBF { unsigned short us[16]; uint4 q[2]; v16bf v; };
union F8     { v8f v; float f[8]; };

#define AS1 __attribute__((address_space(1)))
#define AS3 __attribute__((address_space(3)))

#if defined(__has_builtin)
# if __has_builtin(__builtin_amdgcn_global_load_async_to_lds_b128)
#  define ASYNC_LDS 1
# endif
#endif
#ifndef ASYNC_LDS
# define ASYNC_LDS 0
#endif

__device__ __forceinline__ unsigned short f2bf(float x) {
    unsigned int u = __float_as_uint(x);
    u = (u + 0x7FFFu + ((u >> 16) & 1u)) >> 16;
    return (unsigned short)u;
}
__device__ __forceinline__ float bf2f(unsigned short h) {
    return __uint_as_float(((unsigned int)h) << 16);
}
__device__ __forceinline__ float sigmoidf_(float x) {
    return 1.0f / (1.0f + __expf(-x));
}
__device__ __forceinline__ v8f wmma_bf16(const PackBF& a, const PackBF& b, v8f c) {
    return __builtin_amdgcn_wmma_f32_16x16x32_bf16(
        false, a.v, false, b.v, (short)0, c, false, false);
}

// 16-byte global -> LDS copy; async DMA path when the builtin exists.
__device__ __forceinline__ void async_b128(const void* g, void* l) {
#if ASYNC_LDS
    __builtin_amdgcn_global_load_async_to_lds_b128(
        (AS1 v4i*)(unsigned long long)g,
        (AS3 v4i*)(unsigned int)(unsigned long long)l,
        0, 0);
#else
    *(uint4*)l = *(const uint4*)g;
#endif
}
__device__ __forceinline__ void wait_async() {
#if defined(__has_builtin)
# if __has_builtin(__builtin_amdgcn_s_wait_asynccnt)
    __builtin_amdgcn_s_wait_asynccnt(0);
# else
    asm volatile("s_wait_asynccnt 0" ::: "memory");
# endif
#else
    asm volatile("s_wait_asynccnt 0" ::: "memory");
#endif
}

// ---------------------------------------------------------------------------
// Embedding: x + scale * sinusoidal PE; emit fp32 (residual) + bf16 (GEMM in)
// ---------------------------------------------------------------------------
__global__ void embed_pe(const float* __restrict__ src, const float* __restrict__ scale,
                         int L, float* __restrict__ e, unsigned short* __restrict__ ebf) {
    long idx = (long)blockIdx.x * 256 + threadIdx.x;
    long total = (long)B_ * L * H_;
    if (idx >= total) return;
    int c   = (int)(idx & 255);
    int pos = (int)((idx >> 8) % L);
    float i2  = (float)(c & ~1);                       // 2t
    float div = __expf(-i2 * (9.210340371976184f / 256.0f)); // ln(1e4)
    float ang = (float)pos * div;
    float pe  = (c & 1) ? __cosf(ang) : __sinf(ang);
    float v = src[idx] + scale[0] * pe;
    e[idx]   = v;
    ebf[idx] = f2bf(v);
}

// ---------------------------------------------------------------------------
// Weight transpose: w [din=256][dout=256] fp32 -> wt bf16 [dout][din]
// ---------------------------------------------------------------------------
__global__ void transpose_w(const float* __restrict__ w, unsigned short* __restrict__ wt) {
    int idx = blockIdx.x * 256 + threadIdx.x;   // 65536 total
    int n = idx >> 8, k = idx & 255;
    wt[idx] = f2bf(w[k * 256 + n]);
}

// ---------------------------------------------------------------------------
// LDS-staged 256x256 GEMM: Y[M,256] = Xbf16[M,256] @ WT[256(N)][256(K)]^T + bias
// 256 threads (8 waves) per 64x256 output block. X/W K-chunks staged via
// global_load_async_to_lds_b128; each wave runs 8 WMMAs per K-step from LDS.
// ---------------------------------------------------------------------------
__global__ void __launch_bounds__(256)
gemm256_bf16(const unsigned short* __restrict__ X, const unsigned short* __restrict__ WT,
             const float* __restrict__ bias, float* __restrict__ Y, int M) {
    __shared__ unsigned short Xs[64][64];    // 8 KB
    __shared__ unsigned short Ws[256][64];   // 32 KB
    int tid = threadIdx.x, lane = tid & 31, wave = tid >> 5;
    int half = lane >> 4, lq = lane & 15;
    int m0 = blockIdx.x * 64;
    int mi = wave >> 1, nh = wave & 1;

    F8 acc[8];
    for (int t = 0; t < 8; ++t) acc[t].v = (v8f){};

    int xr = tid >> 2, xs = (tid & 3) * 16;  // X staging: row, 16-elem segment
    for (int k0 = 0; k0 < 256; k0 += 64) {
        const unsigned short* xrow = X + (size_t)(m0 + xr) * 256 + k0 + xs;
        async_b128(xrow,     &Xs[xr][xs]);
        async_b128(xrow + 8, &Xs[xr][xs + 8]);
        const unsigned short* wrow = WT + (size_t)tid * 256 + k0;
        for (int i = 0; i < 8; ++i)
            async_b128(wrow + i * 8, &Ws[tid][i * 8]);
        wait_async();
        __syncthreads();

        for (int ks = 0; ks < 64; ks += 32) {
            PackBF a;
            a.q[0] = *(const uint4*)&Xs[mi * 16 + lq][ks + half * 8];
            a.q[1] = *(const uint4*)&Xs[mi * 16 + lq][ks + 16 + half * 8];
            for (int nt = 0; nt < 8; ++nt) {
                int n = nh * 128 + nt * 16;
                PackBF b;
                b.q[0] = *(const uint4*)&Ws[n + lq][ks + half * 8];
                b.q[1] = *(const uint4*)&Ws[n + lq][ks + 16 + half * 8];
                acc[nt].v = wmma_bf16(a, b, acc[nt].v);
            }
        }
        __syncthreads();
    }
    int mrow = m0 + mi * 16;
    for (int nt = 0; nt < 8; ++nt) {
        int col = nh * 128 + nt * 16 + lq;
        float bv = bias[col];
        for (int r = 0; r < 8; ++r)
            Y[(size_t)(mrow + r + 8 * half) * 256 + col] = acc[nt].f[r] + bv;
    }
}

// ---------------------------------------------------------------------------
// Pack Q~ = [a*Qd , (1-a)*Kd] * 1/sqrt(dk)  -> bf16 [B*H][DLP_][64]
// ---------------------------------------------------------------------------
__global__ void pack_qt(const float* __restrict__ Qd_f, const float* __restrict__ Kd_f,
                        const float* __restrict__ alpha, unsigned short* __restrict__ Qt) {
    long idx = (long)blockIdx.x * 256 + threadIdx.x;   // B*H*DLP_*64
    int t = (int)(idx & 63);
    long r = idx >> 6;
    int i  = (int)(r % DLP_);
    int bh = (int)(r / DLP_);
    int b = bh >> 3, h = bh & 7;
    float a = sigmoidf_(alpha[0]);
    float s = ((t < 32) ? a : (1.0f - a)) * INV_SQRT_DK;
    float v = 0.0f;
    if (i < DL_) {
        const float* src = (t < 32) ? Qd_f : Kd_f;
        v = src[(size_t)(b * DL_ + i) * 256 + h * 32 + (t & 31)] * s;
    }
    Qt[idx] = f2bf(v);
}

// K~ = [Kp , Qp] -> bf16 [B*H][PLP_][64]
__global__ void pack_kt(const float* __restrict__ Kp_f, const float* __restrict__ Qp_f,
                        unsigned short* __restrict__ Kt) {
    long idx = (long)blockIdx.x * 256 + threadIdx.x;   // B*H*PLP_*64
    int t = (int)(idx & 63);
    long r = idx >> 6;
    int j  = (int)(r % PLP_);
    int bh = (int)(r / PLP_);
    int b = bh >> 3, h = bh & 7;
    float v = 0.0f;
    if (j < PL_) {
        const float* src = (t < 32) ? Kp_f : Qp_f;
        v = src[(size_t)(b * PL_ + j) * 256 + h * 32 + (t & 31)];
    }
    Kt[idx] = f2bf(v);
}

// V transpose per head: V_f[B*L][256] -> VT bf16 [B*H][32][LP]
__global__ void pack_vt(const float* __restrict__ V_f, unsigned short* __restrict__ VT,
                        int L, int LP) {
    long idx = (long)blockIdx.x * 256 + threadIdx.x;   // B*H*32*LP
    int j = (int)(idx % LP);
    long r = idx / LP;
    int d  = (int)(r % 32);
    int bh = (int)(r / 32);
    int b = bh >> 3, h = bh & 7;
    float v = (j < L) ? V_f[(size_t)(b * L + j) * 256 + h * 32 + d] : 0.0f;
    VT[idx] = f2bf(v);
}

// ---------------------------------------------------------------------------
// Attention pass 1: per (b,h,16-row drug tile):
//   logits via WMMA (K=64) into LDS, full softmax in LDS,
//   ctx_d = P @ Vp via WMMA from LDS, normalized P^T spilled bf16.
// ---------------------------------------------------------------------------
#define NTI_ (DLP_ / 16)   // 20
__global__ void __launch_bounds__(128)
attn_fwd(const unsigned short* __restrict__ Qt, const unsigned short* __restrict__ Kt,
         const unsigned short* __restrict__ VpT,
         unsigned short* __restrict__ PT, unsigned short* __restrict__ ctx_dbf) {
    __shared__ unsigned short S[16][PLP_];   // 32 KB bf16 logits -> exp values
    __shared__ float red[16][8];
    __shared__ float rowm[16];
    __shared__ float rowsum[16];
    __shared__ float cacc[2][16][16];

    int tid = threadIdx.x, lane = tid & 31, wave = tid >> 5;
    int wg = blockIdx.x;
    int it = wg % NTI_;
    int bh = wg / NTI_;
    int b = bh >> 3, h = bh & 7;
    int i0 = it * 16;
    int half = lane >> 4, lq = lane & 15;

    // ---- phase 1: logits -------------------------------------------------
    const unsigned short* arow = Qt + ((size_t)bh * DLP_ + i0 + lq) * 64;
    for (int jt = wave; jt < PLP_ / 16; jt += 4) {
        int j0 = jt * 16;
        const unsigned short* brow = Kt + ((size_t)bh * PLP_ + j0 + lq) * 64;
        __builtin_prefetch(brow + 64 * 64, 0, 1);   // next j-tile for this wave
        F8 acc; acc.v = (v8f){};
        for (int k0 = 0; k0 < 64; k0 += 32) {
            PackBF a, bb;
            a.q[0]  = *(const uint4*)(arow + k0 + half * 8);
            a.q[1]  = *(const uint4*)(arow + k0 + 16 + half * 8);
            bb.q[0] = *(const uint4*)(brow + k0 + half * 8);
            bb.q[1] = *(const uint4*)(brow + k0 + 16 + half * 8);
            acc.v = wmma_bf16(a, bb, acc.v);
        }
        int j = j0 + lq;
        for (int r = 0; r < 8; ++r)
            S[r + 8 * half][j] = f2bf((j < PL_) ? acc.f[r] : -1e30f);
    }
    __syncthreads();

    // ---- phase 2: softmax stats -----------------------------------------
    int row = tid & 15, seg = tid >> 4;
    float lmax = -1e30f;
    for (int j = seg * 128; j < seg * 128 + 128; ++j)
        lmax = fmaxf(lmax, bf2f(S[row][j]));
    red[row][seg] = lmax;
    __syncthreads();
    if (tid < 16) {
        float m = red[tid][0];
        for (int q = 1; q < 8; ++q) m = fmaxf(m, red[tid][q]);
        rowm[tid] = m;
    }
    __syncthreads();
    float m = rowm[row];
    float ls = 0.0f;
    for (int j = seg * 128; j < seg * 128 + 128; ++j) {
        float e = __expf(bf2f(S[row][j]) - m);
        S[row][j] = f2bf(e);
        ls += e;
    }
    red[row][seg] = ls;
    __syncthreads();
    if (tid < 16) {
        float s = 0.0f;
        for (int q = 0; q < 8; ++q) s += red[tid][q];
        rowsum[tid] = s;
    }
    for (int idx = tid; idx < 512; idx += 128) ((float*)cacc)[idx] = 0.0f;
    __syncthreads();

    // ---- phase 3a: spill normalized P^T ---------------------------------
    for (int idx = tid; idx < 16 * PLP_; idx += 128) {
        int j = idx >> 4, ii = idx & 15;
        int i = i0 + ii;
        float p = (i < DL_) ? bf2f(S[ii][j]) * (1.0f / rowsum[ii]) : 0.0f;
        PT[((size_t)bh * PLP_ + j) * DLP_ + i] = f2bf(p);
    }

    // ---- phase 3b: ctx_d = P @ Vp  (WMMA, A gathered from LDS) ----------
    float inva = 1.0f / rowsum[lq];
    F8 c0, c1; c0.v = (v8f){}; c1.v = (v8f){};
    for (int jt = wave; jt < PLP_ / 32; jt += 4) {
        int j0 = jt * 32;
        PackBF a;
        for (int e = 0; e < 16; ++e) {
            int k = (e < 8) ? (half * 8 + e) : (16 + half * 8 + (e - 8));
            a.us[e] = f2bf(bf2f(S[lq][j0 + k]) * inva);
        }
        PackBF b0, b1;
        const unsigned short* v0 = VpT + (size_t)(bh * 32 + lq)      * PLP_ + j0 + half * 8;
        const unsigned short* v1 = VpT + (size_t)(bh * 32 + 16 + lq) * PLP_ + j0 + half * 8;
        b0.q[0] = *(const uint4*)v0; b0.q[1] = *(const uint4*)(v0 + 16);
        b1.q[0] = *(const uint4*)v1; b1.q[1] = *(const uint4*)(v1 + 16);
        c0.v = wmma_bf16(a, b0, c0.v);
        c1.v = wmma_bf16(a, b1, c1.v);
    }
    for (int r = 0; r < 8; ++r) {
        atomicAdd(&cacc[0][r + 8 * half][lq], c0.f[r]);
        atomicAdd(&cacc[1][r + 8 * half][lq], c1.f[r]);
    }
    __syncthreads();
    for (int idx = tid; idx < 512; idx += 128) {
        int nh = idx >> 8, rem = idx & 255, ii = rem >> 4, n = rem & 15;
        int i = i0 + ii;
        if (i < DL_)
            ctx_dbf[(size_t)(b * DL_ + i) * 256 + h * 32 + nh * 16 + n] = f2bf(cacc[nh][ii][n]);
    }
}

// ---------------------------------------------------------------------------
// Attention pass 2: per (b,h,16-col tile): ctx_p = P^T @ Vd (WMMA) + column means
// ---------------------------------------------------------------------------
__global__ void __launch_bounds__(128)
attn_ctx_p(const unsigned short* __restrict__ PT, const unsigned short* __restrict__ VdT,
           unsigned short* __restrict__ ctx_pbf, float* __restrict__ colmean) {
    __shared__ float cacc[2][16][16];
    __shared__ float red[16][8];
    int tid = threadIdx.x, lane = tid & 31, wave = tid >> 5;
    int wg = blockIdx.x;
    int jt = wg % 63;          // j tiles covering [0,1008) ⊇ [0,PL)
    int bh = wg / 63;
    int b = bh >> 3, h = bh & 7;
    int j0 = jt * 16;
    int half = lane >> 4, lq = lane & 15;

    for (int idx = tid; idx < 512; idx += 128) ((float*)cacc)[idx] = 0.0f;
    __syncthreads();

    F8 c0, c1; c0.v = (v8f){}; c1.v = (v8f){};
    const unsigned short* arow = PT + ((size_t)bh * PLP_ + j0 + lq) * DLP_;
    for (int ks = wave; ks < DLP_ / 32; ks += 4) {
        int k0 = ks * 32;
        PackBF a, b0, b1;
        a.q[0] = *(const uint4*)(arow + k0 + half * 8);
        a.q[1] = *(const uint4*)(arow + k0 + 16 + half * 8);
        const unsigned short* v0 = VdT + (size_t)(bh * 32 + lq)      * DLP_ + k0 + half * 8;
        const unsigned short* v1 = VdT + (size_t)(bh * 32 + 16 + lq) * DLP_ + k0 + half * 8;
        b0.q[0] = *(const uint4*)v0; b0.q[1] = *(const uint4*)(v0 + 16);
        b1.q[0] = *(const uint4*)v1; b1.q[1] = *(const uint4*)(v1 + 16);
        c0.v = wmma_bf16(a, b0, c0.v);
        c1.v = wmma_bf16(a, b1, c1.v);
    }
    for (int r = 0; r < 8; ++r) {
        atomicAdd(&cacc[0][r + 8 * half][lq], c0.f[r]);
        atomicAdd(&cacc[1][r + 8 * half][lq], c1.f[r]);
    }
    // column sums of P (for mean over i)
    int row = tid & 15, seg = tid >> 4;
    float s = 0.0f;
    const unsigned short* prow = PT + ((size_t)bh * PLP_ + j0 + row) * DLP_;
    for (int i = seg * (DLP_ / 8); i < (seg + 1) * (DLP_ / 8); ++i) s += bf2f(prow[i]);
    red[row][seg] = s;
    __syncthreads();
    if (tid < 16) {
        float t = 0.0f;
        for (int q = 0; q < 8; ++q) t += red[tid][q];
        int j = j0 + tid;
        if (j < PL_) colmean[((size_t)b * PLP_ + j) * 8 + h] = t / (float)DL_;
    }
    for (int idx = tid; idx < 512; idx += 128) {
        int nh = idx >> 8, rem = idx & 255, jj = rem >> 4, n = rem & 15;
        int j = j0 + jj;
        if (j < PL_)
            ctx_pbf[(size_t)(b * PL_ + j) * 256 + h * 32 + nh * 16 + n] = f2bf(cacc[nh][jj][n]);
    }
}

// ---------------------------------------------------------------------------
// drug_attn is a constant (softmax row-mean == 1/PL): sigmoid(sum_h w/PL + b)
// ---------------------------------------------------------------------------
__global__ void dattn_const(const float* __restrict__ w, const float* __restrict__ bias,
                            float* __restrict__ dattn) {
    int c = threadIdx.x;
    float t = 0.0f;
    for (int h = 0; h < HEADS_; ++h) t += w[h * 256 + c];
    dattn[c] = sigmoidf_(bias[c] + t * (1.0f / (float)PL_));
}

// prot_attn = sigmoid(colmean @ w_fc_pd + b), K=8
__global__ void pattn_fc(const float* __restrict__ colmean, const float* __restrict__ w,
                         const float* __restrict__ bias, float* __restrict__ pattn) {
    int c = threadIdx.x;
    int rj = blockIdx.x;            // b*PL + j
    int b = rj / PL_, j = rj % PL_;
    float s = bias[c];
    for (int h = 0; h < HEADS_; ++h)
        s += colmean[((size_t)b * PLP_ + j) * 8 + h] * w[h * 256 + c];
    pattn[(size_t)rj * 256 + c] = sigmoidf_(s);
}

// ---------------------------------------------------------------------------
// Final combines (residual + gated context * attention gate)
// ---------------------------------------------------------------------------
__global__ void combine_d(const float* __restrict__ de, const float* __restrict__ Od,
                          const float* __restrict__ Gd, const float* __restrict__ dattn,
                          float* __restrict__ out) {
    long idx = (long)blockIdx.x * 256 + threadIdx.x;   // B*DL*256
    int c = (int)(idx & 255);
    long r = idx >> 8;
    int b = (int)(r / DL_), i = (int)(r % DL_);
    float v = de[idx] + Od[idx] * sigmoidf_(Gd[idx]) * dattn[c];
    out[((size_t)b * NROW_ + i) * 256 + c] = v;
}
__global__ void combine_p(const float* __restrict__ pe, const float* __restrict__ Op,
                          const float* __restrict__ Gp, const float* __restrict__ pattn,
                          float* __restrict__ out) {
    long idx = (long)blockIdx.x * 256 + threadIdx.x;   // B*PL*256
    int c = (int)(idx & 255);
    long r = idx >> 8;
    int b = (int)(r / PL_), j = (int)(r % PL_);
    float v = pe[idx] + Op[idx] * sigmoidf_(Gp[idx]) * pattn[idx];
    out[((size_t)b * NROW_ + DL_ + j) * 256 + c] = v;
}

// ---------------------------------------------------------------------------
extern "C" void kernel_launch(void* const* d_in, const int* in_sizes, int n_in,
                              void* d_out, int out_size, void* d_ws, size_t ws_size,
                              hipStream_t stream) {
    (void)in_sizes; (void)n_in; (void)out_size; (void)ws_size;
    const float* drug    = (const float*)d_in[0];
    const float* prot    = (const float*)d_in[1];
    const float* scale_d = (const float*)d_in[2];
    const float* scale_p = (const float*)d_in[3];
    const float* alpha   = (const float*)d_in[20];
    float* out = (float*)d_out;

    char* w = (char*)d_ws;
    size_t off = 0;
    auto alloc = [&](size_t bytes) -> void* {
        off = (off + 255) & ~(size_t)255;
        void* p = w + off; off += bytes; return p;
    };
    const size_t MD = (size_t)B_ * DL_ * 256;   // 9280*256
    const size_t MP = (size_t)B_ * PL_ * 256;   // 32000*256

    float*          de      = (float*)alloc(MD * 4);
    float*          pe      = (float*)alloc(MP * 4);
    unsigned short* debf    = (unsigned short*)alloc(MD * 2);
    unsigned short* pebf    = (unsigned short*)alloc(MP * 2);
    unsigned short* WT      = (unsigned short*)alloc((size_t)10 * 65536 * 2);
    float*          Qd_f    = (float*)alloc(MD * 4);
    float*          Kd_f    = (float*)alloc(MD * 4);
    float*          Vd_f    = (float*)alloc(MD * 4);
    float*          Kp_f    = (float*)alloc(MP * 4);
    float*          Vp_f    = (float*)alloc(MP * 4);
    float*          Qp_f    = (float*)alloc(MP * 4);
    unsigned short* Qt      = (unsigned short*)alloc((size_t)B_ * HEADS_ * DLP_ * 64 * 2);
    unsigned short* Kt      = (unsigned short*)alloc((size_t)B_ * HEADS_ * PLP_ * 64 * 2);
    unsigned short* VpT     = (unsigned short*)alloc((size_t)B_ * HEADS_ * 32 * PLP_ * 2);
    unsigned short* VdT     = (unsigned short*)alloc((size_t)B_ * HEADS_ * 32 * DLP_ * 2);
    unsigned short* PT      = (unsigned short*)alloc((size_t)B_ * HEADS_ * PLP_ * DLP_ * 2);
    unsigned short* ctx_dbf = (unsigned short*)alloc(MD * 2);
    unsigned short* ctx_pbf = (unsigned short*)alloc(MP * 2);
    float*          colmean = (float*)alloc((size_t)B_ * PLP_ * 8 * 4);
    float*          Od      = (float*)alloc(MD * 4);
    float*          Gd      = (float*)alloc(MD * 4);
    float*          Op      = (float*)alloc(MP * 4);
    float*          Gp      = (float*)alloc(MP * 4);
    float*          pattn   = (float*)alloc(MP * 4);
    float*          dattn   = (float*)alloc(256 * 4);

    // 1. embeddings
    embed_pe<<<(unsigned)(MD / 256), 256, 0, stream>>>(drug, scale_d, DL_, de, debf);
    embed_pe<<<(unsigned)(MP / 256), 256, 0, stream>>>(prot, scale_p, PL_, pe, pebf);

    // 2. weight transposes (slices: qd,kp,vp,qp,kd,vd,outd,outp,gd,gp)
    const int widx[10] = {4, 6, 8, 10, 12, 14, 16, 18, 25, 27};
    for (int s = 0; s < 10; ++s)
        transpose_w<<<256, 256, 0, stream>>>((const float*)d_in[widx[s]], WT + (size_t)s * 65536);

    // 3. projection + gate GEMMs (64-row blocks: 9280/64=145, 32000/64=500)
    gemm256_bf16<<<145, 256, 0, stream>>>(debf, WT + 0 * 65536, (const float*)d_in[5],  Qd_f, 9280);
    gemm256_bf16<<<500, 256, 0, stream>>>(pebf, WT + 1 * 65536, (const float*)d_in[7],  Kp_f, 32000);
    gemm256_bf16<<<500, 256, 0, stream>>>(pebf, WT + 2 * 65536, (const float*)d_in[9],  Vp_f, 32000);
    gemm256_bf16<<<500, 256, 0, stream>>>(pebf, WT + 3 * 65536, (const float*)d_in[11], Qp_f, 32000);
    gemm256_bf16<<<145, 256, 0, stream>>>(debf, WT + 4 * 65536, (const float*)d_in[13], Kd_f, 9280);
    gemm256_bf16<<<145, 256, 0, stream>>>(debf, WT + 5 * 65536, (const float*)d_in[15], Vd_f, 9280);
    gemm256_bf16<<<145, 256, 0, stream>>>(debf, WT + 8 * 65536, (const float*)d_in[26], Gd, 9280);
    gemm256_bf16<<<500, 256, 0, stream>>>(pebf, WT + 9 * 65536, (const float*)d_in[28], Gp, 32000);

    // 4. pack attention operands
    pack_qt<<<(unsigned)((size_t)B_ * HEADS_ * DLP_ * 64 / 256), 256, 0, stream>>>(Qd_f, Kd_f, alpha, Qt);
    pack_kt<<<(unsigned)((size_t)B_ * HEADS_ * PLP_ * 64 / 256), 256, 0, stream>>>(Kp_f, Qp_f, Kt);
    pack_vt<<<(unsigned)((size_t)B_ * HEADS_ * 32 * PLP_ / 256), 256, 0, stream>>>(Vp_f, VpT, PL_, PLP_);
    pack_vt<<<(unsigned)((size_t)B_ * HEADS_ * 32 * DLP_ / 256), 256, 0, stream>>>(Vd_f, VdT, DL_, DLP_);

    // 5. fused attention
    attn_fwd<<<B_ * HEADS_ * NTI_, 128, 0, stream>>>(Qt, Kt, VpT, PT, ctx_dbf);
    attn_ctx_p<<<B_ * HEADS_ * 63, 128, 0, stream>>>(PT, VdT, ctx_pbf, colmean);

    // 6. output projections
    gemm256_bf16<<<145, 256, 0, stream>>>(ctx_dbf, WT + 6 * 65536, (const float*)d_in[17], Od, 9280);
    gemm256_bf16<<<500, 256, 0, stream>>>(ctx_pbf, WT + 7 * 65536, (const float*)d_in[19], Op, 32000);

    // 7. attention-mean FCs
    dattn_const<<<1, 256, 0, stream>>>((const float*)d_in[21], (const float*)d_in[22], dattn);
    pattn_fc<<<B_ * PL_, 256, 0, stream>>>(colmean, (const float*)d_in[23], (const float*)d_in[24], pattn);

    // 8. combine
    combine_d<<<(unsigned)(MD / 256), 256, 0, stream>>>(de, Od, Gd, dattn, out);
    combine_p<<<(unsigned)(MP / 256), 256, 0, stream>>>(pe, Op, Gp, pattn, out);
}
